// SimpleNet_84224308674623
// MI455X (gfx1250) — matrix operand order repaired and checked by reading, not compiled, not run on hardware
//
#include <hip/hip_runtime.h>
#include <hip/hip_bf16.h>

// ---------------------------------------------------------------------------
// Problem dimensions (compile-time)
// ---------------------------------------------------------------------------
#define B_ 32
#define T_ 512
#define IN_ 256
#define H_ 1024
#define P_ 512
#define O_ 256
#define G_ (4 * H_)       // 4096 gate columns
#define RT_ (B_ * T_)     // 16384 rows of xg / hp_seq
#define NBLK_SCAN 32

typedef __attribute__((ext_vector_type(16))) __bf16 v16bf;
typedef __attribute__((ext_vector_type(8)))  __bf16 v8bf;
typedef __attribute__((ext_vector_type(8)))  float  v8f;
typedef __attribute__((ext_vector_type(4)))  float  v4f;
typedef __attribute__((ext_vector_type(4)))  int    v4i;

typedef __attribute__((address_space(1))) v4i* gv4i_p;  // global v4i*
typedef __attribute__((address_space(3))) v4i* lv4i_p;  // LDS v4i*

// ---------------------------------------------------------------------------
// Async global -> LDS copy (CDNA5 GLOBAL_LOAD_ASYNC_TO_LDS_B128, ASYNCcnt).
// Guarded: falls back to a plain vector copy if the builtin is unavailable.
// ---------------------------------------------------------------------------
#if defined(__has_builtin)
#if __has_builtin(__builtin_amdgcn_global_load_async_to_lds_b128)
#define ASYNC_LDS 1
#endif
#endif
#ifndef ASYNC_LDS
#define ASYNC_LDS 0
#endif

__device__ __forceinline__ void copy16_g2l(const float* g, float* l) {
#if ASYNC_LDS
  float* gnc = const_cast<float*>(g);
  __builtin_amdgcn_global_load_async_to_lds_b128(
      (gv4i_p)(v4i*)gnc, (lv4i_p)(v4i*)l, 0, 0);
#else
  *(v4f*)l = *(const v4f*)g;
#endif
}

__device__ __forceinline__ void wait_async_done() {
#if ASYNC_LDS
#if __has_builtin(__builtin_amdgcn_s_wait_asynccnt)
  __builtin_amdgcn_s_wait_asynccnt(0);
#else
  asm volatile("s_wait_asynccnt 0x0" ::: "memory");
#endif
#endif
}

// ---------------------------------------------------------------------------
// WMMA helpers (CDNA5 wave32, v_wmma_f32_16x16x32_bf16)
//
// A fragment (16x32, MxK): lane holds row M=lane%16; half=lane/16;
//   values: k = kc + 16*half + {0..15}
// B fragment (32x16, KxN): lane holds col N=lane%16; half=lane/16;
//   values: k = kc + 16*half + {0..15}
// C/D (16x16 f32): lane -> N=lane%16, element v -> M = v + 8*(lane/16)
// ---------------------------------------------------------------------------
__device__ __forceinline__ v8f wmma_bf16(v16bf a, v16bf b, v8f c) {
  return __builtin_amdgcn_wmma_f32_16x16x32_bf16(
      /*neg_a=*/false, a, /*neg_b=*/false, b,
      /*c_mod=*/(short)0, c, /*reuse_a=*/false, /*reuse_b=*/false);
}

// rowptr already includes this lane's row (row-major, k-contiguous storage)
__device__ __forceinline__ v16bf load_a_frag(const __bf16* rowptr, int kc) {
  const int half = (threadIdx.x & 31) >> 4;
  const __bf16* p = rowptr + kc + half * 16;
  v8bf lo = *(const v8bf*)(p);
  v8bf hi = *(const v8bf*)(p + 8);
  return __builtin_shufflevector(lo, hi, 0, 1, 2, 3, 4, 5, 6, 7,
                                 8, 9, 10, 11, 12, 13, 14, 15);
}

// Bt stored [n][k] (k-contiguous), leading dim ldb (elements)
__device__ __forceinline__ v16bf load_b_frag(const __bf16* Bt, int ldb,
                                             int nbase, int kc) {
  const int lane = threadIdx.x & 31;
  const int half = lane >> 4;
  const __bf16* p = Bt + (size_t)(nbase + (lane & 15)) * ldb + kc + half * 16;
  v8bf lo = *(const v8bf*)(p);
  v8bf hi = *(const v8bf*)(p + 8);
  return __builtin_shufflevector(lo, hi, 0, 1, 2, 3, 4, 5, 6, 7,
                                 8, 9, 10, 11, 12, 13, 14, 15);
}

__device__ __forceinline__ float sigmoidf_(float x) {
  return 1.0f / (1.0f + __expf(-x));
}

// ---------------------------------------------------------------------------
// Prep: fp32 -> bf16 conversions / transposes
// ---------------------------------------------------------------------------
__global__ void k_convert_bf16(const float* __restrict__ src,
                               __bf16* __restrict__ dst, int n) {
  int i = blockIdx.x * blockDim.x + threadIdx.x;
  if (i < n) dst[i] = (__bf16)src[i];
}

// src[rows][cols] -> dst[cols][rows]
__global__ void k_transpose_bf16(const float* __restrict__ src,
                                 __bf16* __restrict__ dst, int rows, int cols) {
  int i = blockIdx.x * blockDim.x + threadIdx.x;
  if (i < rows * cols) {
    int r = i / cols, c = i - r * cols;
    dst[(size_t)c * rows + r] = (__bf16)src[i];
  }
}

// ---------------------------------------------------------------------------
// Kernel: xg[t*32+b][g] = sum_i inputs[b][t][i] * W[i][g] + bias[g]
// A = in_bf16 [b][t][i] (rows remapped), B = Wt_bf16 [g][i]
// block = 256 thr (8 waves), tile 32x64, grid (G/64, RT/32)
// ---------------------------------------------------------------------------
__global__ void __launch_bounds__(256) k_xg_gemm(
    const __bf16* __restrict__ Abf, const __bf16* __restrict__ Wt,
    const float* __restrict__ bias, float* __restrict__ xg) {
  const int lane = threadIdx.x & 31;
  const int w = threadIdx.x >> 5;
  const int half = lane >> 4;
  const int mt = w >> 2, nt = w & 3;
  const int rbase = blockIdx.y * 32 + mt * 16;
  const int nbase = blockIdx.x * 64 + nt * 16;

  const int r = rbase + (lane & 15);        // row = t*32 + b
  const int bb = r & 31, tt = r >> 5;
  const __bf16* arow = Abf + ((size_t)bb * T_ + tt) * IN_;

  v8f acc;
#pragma unroll
  for (int v = 0; v < 8; ++v) acc[v] = 0.0f;

#pragma unroll
  for (int kc = 0; kc < IN_; kc += 32) {
    v16bf a = load_a_frag(arow, kc);
    v16bf b = load_b_frag(Wt, IN_, nbase, kc);
    acc = wmma_bf16(a, b, acc);
  }

  const int n = nbase + (lane & 15);
  const float bv = bias[n];
#pragma unroll
  for (int v = 0; v < 8; ++v) {
    int rr = rbase + v + 8 * half;
    xg[(size_t)rr * G_ + n] = acc[v] + bv;
  }
}

// ---------------------------------------------------------------------------
// Persistent recurrent scan: 32 blocks x 512 threads (16 waves)
// block owns 32 hidden units. Grid sync via atomic counter + s_sleep spin.
// xg[t] slabs are double-buffered in LDS via async global->LDS loads.
// ---------------------------------------------------------------------------
__device__ __forceinline__ void grid_barrier(unsigned* cnt,
                                             volatile unsigned* rel,
                                             unsigned gen) {
  __syncthreads();
  if (threadIdx.x == 0) {
    __threadfence();
    unsigned prev = atomicAdd(cnt, 1u);
    if (prev == (unsigned)NBLK_SCAN * gen - 1u) {
      __threadfence();
      *rel = gen;
    } else {
      while (*rel < gen) __builtin_amdgcn_s_sleep(2);
    }
    __threadfence();
  }
  __syncthreads();
}

// issue 16B async transfers for this block's xg slab at step t
// dst points to xgb[buf] laid out as [4 gates][32 batch][32 cols]
__device__ __forceinline__ void prefetch_xg(const float* __restrict__ xg,
                                            int t, int hbase, float* dst,
                                            int tid) {
#pragma unroll
  for (int rep = 0; rep < 2; ++rep) {
    int e = tid + rep * 512;                 // 0..1023, 16B each
    int m = e >> 5;                          // batch row 0..31
    int gate = (e >> 3) & 3;                 // 0..3
    int chunk = e & 7;                       // 0..7 (4 floats each)
    const float* g =
        xg + ((size_t)t * B_ + m) * G_ + gate * H_ + hbase + chunk * 4;
    float* l = dst + ((gate * 32 + m) * 32) + chunk * 4;
    copy16_g2l(g, l);
  }
}

__global__ void __launch_bounds__(512, 1) k_lstm_scan(
    const float* __restrict__ xg,        // [T*32][4096]
    const __bf16* __restrict__ Rt,       // [4096][512]
    const __bf16* __restrict__ Wpt,      // [512][1024]
    const float* __restrict__ peep,      // [3*1024]
    __bf16* __restrict__ hp_all,         // [(T+1)][32][512]
    __bf16* __restrict__ hbuf,           // [32][1024]
    unsigned* bar_cnt, volatile unsigned* bar_rel) {
  // zlds (phase 1-2) and plds (phase 3) never live at the same time: union.
  __shared__ float zp[4096];             // 16 KB
  __shared__ float clds[32][32];         // 4 KB cell state [batch][hb]
  __shared__ float xgb[2][4 * 32 * 32];  // 32 KB double-buffered xg slab
#define ZLDS(g, b, hb) zp[(((g)*32 + (b)) * 32) + (hb)]
#define PLDS(q, m, n) zp[(((q)*16 + (m)) * 16) + (n)]

  const int tid = threadIdx.x;
  const int lane = tid & 31;
  const int w = tid >> 5;
  const int half = lane >> 4;
  const int blk = blockIdx.x;
  const int hbase = blk * 32;

  for (int e = tid; e < 32 * 32; e += 512) clds[e >> 5][e & 31] = 0.0f;

  // kick off the async fill of the t=0 slab
  prefetch_xg(xg, 0, hbase, xgb[0], tid);

  unsigned gen = 0;
  for (int t = 0; t < T_; ++t) {
    // xg slab for this step must be fully landed in LDS
    wait_async_done();
    __syncthreads();
    if (t + 1 < T_) prefetch_xg(xg, t + 1, hbase, xgb[(t + 1) & 1], tid);
    const float* xgt = xgb[t & 1];

    // ---- Phase 1: gate GEMM  z = xg_t + hp @ R  (this block's 128 cols) ----
    {
      const int gate = w >> 2;
      const int mt = (w >> 1) & 1;
      const int nt = w & 1;
      const int gcol = gate * H_ + hbase + nt * 16;
      const int mb = mt * 16;
      const int n = lane & 15;
      const __bf16* A = hp_all + (size_t)t * B_ * P_;
      const __bf16* arow = A + (size_t)(mb + n) * P_;

      v8f acc;
#pragma unroll
      for (int v = 0; v < 8; ++v)
        acc[v] = xgt[((gate * 32 + (mb + v + 8 * half)) * 32) + nt * 16 + n];

#pragma unroll 4
      for (int kc = 0; kc < P_; kc += 32) {
        v16bf a = load_a_frag(arow, kc);
        v16bf b = load_b_frag(Rt, P_, gcol, kc);
        acc = wmma_bf16(a, b, acc);
      }
#pragma unroll
      for (int v = 0; v < 8; ++v)
        ZLDS(gate, mb + v + 8 * half, nt * 16 + n) = acc[v];
    }
    __syncthreads();

    // ---- Phase 2: gate nonlinearities + cell update ------------------------
    for (int e = tid; e < B_ * 32; e += 512) {
      const int b = e >> 5, hb = e & 31;
      const int hg = hbase + hb;
      const float zi = ZLDS(0, b, hb);
      const float zf = ZLDS(1, b, hb);
      const float zg = ZLDS(2, b, hb);
      const float zo = ZLDS(3, b, hb);
      const float c = clds[b][hb];
      const float ig = sigmoidf_(zi + c * peep[hg]);
      const float fg = sigmoidf_(zf + c * peep[H_ + hg]);
      const float gg = tanhf(zg);
      const float cn = fg * c + ig * gg;
      const float og = sigmoidf_(zo + cn * peep[2 * H_ + hg]);
      const float h = og * tanhf(cn);
      clds[b][hb] = cn;
      hbuf[(size_t)b * H_ + hg] = (__bf16)h;
    }
    grid_barrier(bar_cnt, bar_rel, ++gen);

    // ---- Phase 3: projection hp = h @ W_proj (this block's 16 cols) --------
    {
      const int mt = w & 1;
      const int kq = w >> 1;             // 0..7, each does 128 of K=1024
      const int mb = mt * 16;
      const int nb = blk * 16;
      const __bf16* arow = hbuf + (size_t)(mb + (lane & 15)) * H_;

      v8f acc;
#pragma unroll
      for (int v = 0; v < 8; ++v) acc[v] = 0.0f;

      const int k0 = kq * 128;
#pragma unroll
      for (int kc = k0; kc < k0 + 128; kc += 32) {
        v16bf a = load_a_frag(arow, kc);
        v16bf b = load_b_frag(Wpt, H_, nb, kc);
        acc = wmma_bf16(a, b, acc);
      }
#pragma unroll
      for (int v = 0; v < 8; ++v)
        PLDS(kq * 2 + mt, v + 8 * half, lane & 15) = acc[v];
    }
    __syncthreads();
    {
      // reduce the 8 K-partials, write hp_all[t+1]
      const int mt2 = tid >> 8;          // 0..1
      const int m2 = (tid >> 4) & 15;    // 0..15
      const int n2 = tid & 15;
      float s = 0.0f;
#pragma unroll
      for (int kq2 = 0; kq2 < 8; ++kq2) s += PLDS(kq2 * 2 + mt2, m2, n2);
      const int b = mt2 * 16 + m2;
      hp_all[(size_t)(t + 1) * B_ * P_ + (size_t)b * P_ + blk * 16 + n2] =
          (__bf16)s;
    }
    grid_barrier(bar_cnt, bar_rel, ++gen);
  }
#undef ZLDS
#undef PLDS
}

// ---------------------------------------------------------------------------
// Kernel: out[b][t][o] = hp_seq[t*32+b][:] . lin_w[o][:] + lin_b[o]
// Hp = hp_all + 32*512 (rows r = t*32+b are contiguous)
// ---------------------------------------------------------------------------
__global__ void __launch_bounds__(256) k_out_gemm(
    const __bf16* __restrict__ Hp, const __bf16* __restrict__ Lw,
    const float* __restrict__ lb, float* __restrict__ out) {
  const int lane = threadIdx.x & 31;
  const int w = threadIdx.x >> 5;
  const int half = lane >> 4;
  const int mt = w >> 2, nt = w & 3;
  const int rbase = blockIdx.y * 32 + mt * 16;
  const int nbase = blockIdx.x * 64 + nt * 16;
  const __bf16* arow = Hp + (size_t)(rbase + (lane & 15)) * P_;

  v8f acc;
#pragma unroll
  for (int v = 0; v < 8; ++v) acc[v] = 0.0f;

#pragma unroll 4
  for (int kc = 0; kc < P_; kc += 32) {
    v16bf a = load_a_frag(arow, kc);
    v16bf b = load_b_frag(Lw, P_, nbase, kc);
    acc = wmma_bf16(a, b, acc);
  }

  const int n = nbase + (lane & 15);
  const float bv = lb[n];
#pragma unroll
  for (int v = 0; v < 8; ++v) {
    const int r = rbase + v + 8 * half;      // r = t*32 + b
    const int tt = r >> 5, bb = r & 31;
    out[((size_t)bb * T_ + tt) * O_ + n] = acc[v] + bv;
  }
}

// ---------------------------------------------------------------------------
// Host launch
// ---------------------------------------------------------------------------
extern "C" void kernel_launch(void* const* d_in, const int* in_sizes, int n_in,
                              void* d_out, int out_size, void* d_ws,
                              size_t ws_size, hipStream_t stream) {
  const float* inputs = (const float*)d_in[0];  // [32][512][256]
  const float* W      = (const float*)d_in[1];  // [256][4096]
  const float* R      = (const float*)d_in[2];  // [512][4096]
  const float* Wproj  = (const float*)d_in[3];  // [1024][512]
  const float* peep   = (const float*)d_in[4];  // [3072]
  const float* bias   = (const float*)d_in[5];  // [4096]
  const float* lin_w  = (const float*)d_in[6];  // [256][512]
  const float* lin_b  = (const float*)d_in[7];  // [256]
  float* out = (float*)d_out;

  // Workspace carve-up (256B aligned)
  char* ws = (char*)d_ws;
  size_t off = 0;
  auto carve = [&](size_t bytes) {
    char* p = ws + off;
    off = (off + bytes + 255) & ~(size_t)255;
    return p;
  };
  float*  xg     = (float*)carve((size_t)RT_ * G_ * 4);        // 256 MB
  __bf16* in_bf  = (__bf16*)carve((size_t)RT_ * IN_ * 2);      // 8 MB
  __bf16* Wt     = (__bf16*)carve((size_t)G_ * IN_ * 2);       // 2 MB
  __bf16* Rt     = (__bf16*)carve((size_t)G_ * P_ * 2);        // 4 MB
  __bf16* Wpt    = (__bf16*)carve((size_t)P_ * H_ * 2);        // 1 MB
  __bf16* Lw     = (__bf16*)carve((size_t)O_ * P_ * 2);        // 0.25 MB
  __bf16* hp_all = (__bf16*)carve((size_t)(T_ + 1) * B_ * P_ * 2);
  __bf16* hbuf   = (__bf16*)carve((size_t)B_ * H_ * 2);
  unsigned* bar  = (unsigned*)carve(256);

  // Per-launch resets (graph-capturable memset nodes)
  (void)hipMemsetAsync(bar, 0, 256, stream);
  (void)hipMemsetAsync(hp_all, 0, (size_t)B_ * P_ * 2, stream);  // hp(t=0)=0

  // Prep: conversions / transposes to bf16 [n][k] layouts
  {
    int n = RT_ * IN_;
    k_convert_bf16<<<(n + 255) / 256, 256, 0, stream>>>(inputs, in_bf, n);
  }
  {
    int n = IN_ * G_;
    k_transpose_bf16<<<(n + 255) / 256, 256, 0, stream>>>(W, Wt, IN_, G_);
  }
  {
    int n = P_ * G_;
    k_transpose_bf16<<<(n + 255) / 256, 256, 0, stream>>>(R, Rt, P_, G_);
  }
  {
    int n = H_ * P_;
    k_transpose_bf16<<<(n + 255) / 256, 256, 0, stream>>>(Wproj, Wpt, H_, P_);
  }
  {
    int n = O_ * P_;
    k_convert_bf16<<<(n + 255) / 256, 256, 0, stream>>>(lin_w, Lw, n);
  }

  // xg = inputs @ W + b   (WMMA bf16)
  {
    dim3 grid(G_ / 64, RT_ / 32);
    k_xg_gemm<<<grid, 256, 0, stream>>>(in_bf, Wt, bias, xg);
  }

  // Persistent recurrent scan (32 co-resident blocks, grid barriers)
  k_lstm_scan<<<NBLK_SCAN, 512, 0, stream>>>(xg, Rt, Wpt, peep, hp_all, hbuf,
                                             bar, bar + 32);

  // out = hp_seq @ lin_w.T + lin_b   (WMMA bf16)
  {
    dim3 grid(O_ / 64, RT_ / 32);
    k_out_gemm<<<grid, 256, 0, stream>>>(hp_all + (size_t)B_ * P_, Lw, lin_b,
                                         out);
  }
}
